// Net2_2602750181918
// MI455X (gfx1250) — compile-verified, matrix operand order checked by reading, not run on hardware
//
#include <hip/hip_runtime.h>
#include <hip/hip_bf16.h>
#include <stdint.h>

#define NRES 512
#define NN   (NRES * NRES)   // 262144 flattened (i,j) positions
#define CDIM 128

typedef __attribute__((ext_vector_type(16))) __bf16 v16bf;
typedef __attribute__((ext_vector_type(8)))  __bf16 v8bf;
typedef __attribute__((ext_vector_type(8)))  float  v8f;
typedef __attribute__((ext_vector_type(8)))  unsigned short ushort8;
typedef __attribute__((ext_vector_type(4)))  unsigned int u32x4;
typedef __attribute__((ext_vector_type(8)))  int i32x8;
typedef __attribute__((ext_vector_type(4)))  int i32x4;

// ---------- helpers ----------
__device__ __forceinline__ unsigned short f2bf(float f) {
    unsigned int u = __float_as_uint(f);
    u += 0x7FFFu + ((u >> 16) & 1u);          // round-to-nearest-even
    return (unsigned short)(u >> 16);
}
__device__ __forceinline__ float bf2f(unsigned short u) {
    return __uint_as_float(((unsigned int)u) << 16);
}
__device__ __forceinline__ float sigm(float x) { return 1.0f / (1.0f + __expf(-x)); }

__device__ __forceinline__ v16bf ldfrag(const unsigned short* p, int gap) {
    v8bf lo = *(const v8bf*)p;
    v8bf hi = *(const v8bf*)(p + gap);
    return __builtin_shufflevector(lo, hi, 0,1,2,3,4,5,6,7,8,9,10,11,12,13,14,15);
}
__device__ __forceinline__ int frag_k(int lane, int e) {
    int half = lane >> 4;
    return (e < 8) ? (half * 8 + e) : (16 + half * 8 + (e - 8));
}
__device__ __forceinline__ v16bf lda(const unsigned short* s, int rowbase, int rs,
                                     int kofs, int lane) {
    const unsigned short* p = s + (size_t)(rowbase + (lane & 15)) * rs + kofs + (lane >> 4) * 8;
    return ldfrag(p, 16);
}
__device__ __forceinline__ v16bf ldw(const unsigned short* wp, int mat, int nt, int kt, int lane) {
    const unsigned short* p = wp + ((((size_t)mat * 8 + nt) * 4 + kt) * 32 + lane) * 16;
    return ldfrag(p, 8);
}
__device__ __forceinline__ v8f wmma_bf16(v16bf a, v16bf b, v8f c) {
    return __builtin_amdgcn_wmma_f32_16x16x32_bf16(false, a, false, b, (short)0, c, false, false);
}

// ---------- Tensor Data Mover: 2-D tile load, D# built per CDNA5 ISA §8 ----------
// flags = group1 dword0 (data_size / pad控制); tile = til0 x til1 elements,
// tensor dim0 length tdim0 with row stride `stride` (elements).
__device__ __forceinline__ void tdm_load_2d(unsigned lds_off, const void* gaddr,
                                            unsigned flags, unsigned tdim0, unsigned tdim1,
                                            unsigned til0, unsigned til1, unsigned stride) {
    unsigned long long ga = (unsigned long long)(uintptr_t)gaddr;
    u32x4 g0;
    g0[0] = 1u;                                            // count=1 (valid descriptor)
    g0[1] = lds_off;                                       // LDS byte address
    g0[2] = (unsigned)ga;                                  // global_addr[31:0]
    g0[3] = (unsigned)((ga >> 32) & 0x1FFFFFFull) | 0x80000000u; // addr[56:32] | type=2
    i32x8 g1;
    g1[0] = (int)flags;                                    // wg_mask=0, data_size, pad ctl
    g1[1] = (int)((tdim0 & 0xFFFFu) << 16);                // tensor_dim0[15:0] @bit48
    g1[2] = (int)(((tdim0 >> 16) & 0xFFFFu) | ((tdim1 & 0xFFFFu) << 16));
    g1[3] = (int)(((tdim1 >> 16) & 0xFFFFu) | (til0 << 16)); // tile_dim0 @bit112
    g1[4] = (int)(til1 & 0xFFFFu);                         // tile_dim1 (tile_dim2=0)
    g1[5] = (int)stride;                                   // tensor_dim0_stride[31:0]
    g1[6] = 0;
    g1[7] = 0;                                             // tensor_dim1_stride=0 (2-D)
    i32x4 z4 = (i32x4)0;
#if __has_include(<hip/amd_detail/amd_gfx1250_TDM.h>)
    i32x8 z8 = (i32x8)0;
    __builtin_amdgcn_tensor_load_to_lds(g0, g1, z4, z4, z8, 0);
#else
    __builtin_amdgcn_tensor_load_to_lds(g0, g1, z4, z4, 0);
#endif
}

// ---------- kernel 0: pack the 6 [128x128] f32 weights into WMMA B fragments ----------
__global__ void pack_w_kernel(const float* __restrict__ lp, const float* __restrict__ lg,
                              const float* __restrict__ rp, const float* __restrict__ rg,
                              const float* __restrict__ gw, const float* __restrict__ ow,
                              unsigned short* __restrict__ wp) {
    int nt = blockIdx.x, kt = blockIdx.y, mat = blockIdx.z, l = threadIdx.x;
    const float* w = (mat == 0) ? lp : (mat == 1) ? lg : (mat == 2) ? rp
                   : (mat == 3) ? rg : (mat == 4) ? gw : ow;
    unsigned short* dst = wp + ((((size_t)mat * 8 + nt) * 4 + kt) * 32 + l) * 16;
    int n = nt * 16 + (l & 15);
#pragma unroll
    for (int e = 0; e < 16; e++) {
        int k = kt * 32 + frag_k(l, e);
        dst[e] = f2bf(w[k * CDIM + n]);
    }
}

// ---------- kernel 1: input LN + 5 fused GEMMs (lp,lg,rp,rg,gate) ----------
__global__ __launch_bounds__(256)
void proj_kernel(const float* __restrict__ act, const float* __restrict__ mask,
                 const float* __restrict__ g_in, const float* __restrict__ b_in,
                 const float* __restrict__ lp_b, const float* __restrict__ lg_b,
                 const float* __restrict__ rp_b, const float* __restrict__ rg_b,
                 const float* __restrict__ gt_b,
                 const unsigned short* __restrict__ wp,
                 unsigned short* __restrict__ left_t,
                 unsigned short* __restrict__ right_t,
                 unsigned short* __restrict__ gate_t) {
    __shared__ alignas(16) unsigned short sA[128 * CDIM];
    int t = threadIdx.x, wave = t >> 5, lane = t & 31;
    size_t rowbase = (size_t)blockIdx.x * 128;

    { // LayerNorm over C: 2 lanes per row, 64 channels each
        int  rloc = wave * 16 + (lane >> 1);
        int  ch0  = (lane & 1) * 64;
        const float* ar = act + (rowbase + rloc) * CDIM + ch0;
        float vals[64];
        float s = 0.f, ss = 0.f;
#pragma unroll
        for (int i = 0; i < 16; i++) {
            float4 f = ((const float4*)ar)[i];
            vals[4*i+0]=f.x; vals[4*i+1]=f.y; vals[4*i+2]=f.z; vals[4*i+3]=f.w;
            s  += f.x + f.y + f.z + f.w;
            ss += f.x*f.x + f.y*f.y + f.z*f.z + f.w*f.w;
        }
        s  += __shfl_xor(s, 1, 32);
        ss += __shfl_xor(ss, 1, 32);
        float mu   = s * (1.f / CDIM);
        float rstd = rsqrtf(ss * (1.f / CDIM) - mu * mu + 1e-5f);
#pragma unroll
        for (int i = 0; i < 16; i++) {
            float4 g = ((const float4*)(g_in + ch0))[i];
            float4 b = ((const float4*)(b_in + ch0))[i];
            unsigned short* d = &sA[rloc * CDIM + ch0 + 4 * i];
            d[0] = f2bf((vals[4*i+0]-mu)*rstd*g.x + b.x);
            d[1] = f2bf((vals[4*i+1]-mu)*rstd*g.y + b.y);
            d[2] = f2bf((vals[4*i+2]-mu)*rstd*g.z + b.z);
            d[3] = f2bf((vals[4*i+3]-mu)*rstd*g.w + b.w);
        }
    }
    __syncthreads();

    int    mloc = wave * 16;
    size_t pos0 = rowbase + mloc;
    int    lrow = lane & 15, half = lane >> 4;

    float mk[8];
    {
        const float4* mp = (const float4*)(mask + pos0 + half * 8);
        float4 a0 = mp[0], a1 = mp[1];
        mk[0]=a0.x; mk[1]=a0.y; mk[2]=a0.z; mk[3]=a0.w;
        mk[4]=a1.x; mk[5]=a1.y; mk[6]=a1.z; mk[7]=a1.w;
    }

    for (int nt = 0; nt < 8; nt++) {
        int    n     = nt * 16 + lrow;
        size_t obase = (size_t)n * NN + pos0 + half * 8;
        { // LEFT
            v8f ap = {}, ag = {};
#pragma unroll
            for (int kt = 0; kt < 4; kt++) {
                v16bf a = lda(sA, mloc, CDIM, kt * 32, lane);
                ap = wmma_bf16(a, ldw(wp, 0, nt, kt, lane), ap);
                ag = wmma_bf16(a, ldw(wp, 1, nt, kt, lane), ag);
            }
            float pb = lp_b[n], gb = lg_b[n];
            ushort8 o;
#pragma unroll
            for (int r = 0; r < 8; r++) o[r] = f2bf(mk[r] * (ap[r] + pb) * sigm(ag[r] + gb));
            *(ushort8*)(left_t + obase) = o;
        }
        { // RIGHT
            v8f ap = {}, ag = {};
#pragma unroll
            for (int kt = 0; kt < 4; kt++) {
                v16bf a = lda(sA, mloc, CDIM, kt * 32, lane);
                ap = wmma_bf16(a, ldw(wp, 2, nt, kt, lane), ap);
                ag = wmma_bf16(a, ldw(wp, 3, nt, kt, lane), ag);
            }
            float pb = rp_b[n], gb = rg_b[n];
            ushort8 o;
#pragma unroll
            for (int r = 0; r < 8; r++) o[r] = f2bf(mk[r] * (ap[r] + pb) * sigm(ag[r] + gb));
            *(ushort8*)(right_t + obase) = o;
        }
        { // output GATE
            v8f ac = {};
#pragma unroll
            for (int kt = 0; kt < 4; kt++) {
                v16bf a = lda(sA, mloc, CDIM, kt * 32, lane);
                ac = wmma_bf16(a, ldw(wp, 4, nt, kt, lane), ac);
            }
            float gb = gt_b[n];
            ushort8 o;
#pragma unroll
            for (int r = 0; r < 8; r++) o[r] = f2bf(sigm(ac[r] + gb));
            *(ushort8*)(gate_t + obase) = o;
        }
    }
}

// ---------- kernel 2: triangle einsum with TDM double-buffered staging ----------
// x[c][i][j] = sum_k L[c][i][k]*R[c][j][k]; computes D[j][i] so lane stores are
// 8 consecutive j -> contiguous f32. LDS (dynamic, base 0):
//   buf b: L tile at b*16384, R tile at b*16384+8192 (each 128 rows x 32 bf16).
__global__ __launch_bounds__(256)
void tri_kernel(const unsigned short* __restrict__ left_t,
                const unsigned short* __restrict__ right_t,
                float* __restrict__ x_t) {
    extern __shared__ char smem[];
    int t = threadIdx.x, wave = t >> 5, lane = t & 31;
    int c = blockIdx.z;
    size_t ib = (size_t)blockIdx.x * 128, jb = (size_t)blockIdx.y * 128;
    const unsigned short* L = left_t  + (size_t)c * NN + ib * NRES;
    const unsigned short* R = right_t + (size_t)c * NN + jb * NRES;

    int wj = (wave >> 2) * 64;   // 4 j tiles (M side)
    int wi = (wave & 3) * 32;    // 2 i tiles (N side)
    v8f acc[4][2];
    v8f z = {};
#pragma unroll
    for (int a = 0; a < 4; a++)
#pragma unroll
        for (int b = 0; b < 2; b++) acc[a][b] = z;

    const unsigned FL_BF16 = 0x00010000u;   // data_size=2B, no pad, wg_mask=0
    if (wave == 0) {                        // preload k-step 0
        tdm_load_2d(0,    L, FL_BF16, NRES, NRES, 32, 128, NRES);
        tdm_load_2d(8192, R, FL_BF16, NRES, NRES, 32, 128, NRES);
    }

    for (int k = 0; k < 16; k++) {
        int buf = k & 1;
        if (wave == 0) {
            if (k + 1 < 16) {               // prefetch next pair into other buffer
                tdm_load_2d((unsigned)((buf ^ 1) * 16384),        L + (k + 1) * 32,
                            FL_BF16, NRES, NRES, 32, 128, NRES);
                tdm_load_2d((unsigned)((buf ^ 1) * 16384 + 8192), R + (k + 1) * 32,
                            FL_BF16, NRES, NRES, 32, 128, NRES);
                __builtin_amdgcn_s_wait_tensorcnt(2);   // current pair done (in-order)
            } else {
                __builtin_amdgcn_s_wait_tensorcnt(0);
            }
        }
        __syncthreads();                    // tile(k) visible to all waves

        const unsigned short* sL = (const unsigned short*)(smem + buf * 16384);
        const unsigned short* sR = sL + 4096;   // +8192 bytes

        v16bf bfr[2];
        bfr[0] = lda(sL, wi,      32, 0, lane);
        bfr[1] = lda(sL, wi + 16, 32, 0, lane);
#pragma unroll
        for (int jt = 0; jt < 4; jt++) {
            v16bf a = lda(sR, wj + jt * 16, 32, 0, lane);
            acc[jt][0] = wmma_bf16(a, bfr[0], acc[jt][0]);
            acc[jt][1] = wmma_bf16(a, bfr[1], acc[jt][1]);
        }
        __syncthreads();                    // reads done before buf is refilled
    }

    int lrow = lane & 15, half = lane >> 4;
#pragma unroll
    for (int jt = 0; jt < 4; jt++) {
#pragma unroll
        for (int it = 0; it < 2; it++) {
            size_t i = ib + wi + it * 16 + lrow;
            size_t j = jb + wj + jt * 16 + half * 8;
            float* d = x_t + (size_t)c * NN + i * NRES + j;
            v8f v = acc[jt][it];
            ((float4*)d)[0] = make_float4(v[0], v[1], v[2], v[3]);
            ((float4*)d)[1] = make_float4(v[4], v[5], v[6], v[7]);
        }
    }
}

// ---------- kernel 3: LN over I + out GEMM + gate multiply ----------
// TDM loads the [128 c][64 pos] f32 tile with LDS padding (interval 64 dwords,
// amount 1 dword) -> row stride 65 floats, bank-conflict-free column reads.
__global__ __launch_bounds__(256)
void out_kernel(const float* __restrict__ x_t,
                const float* __restrict__ g_c, const float* __restrict__ b_c,
                const float* __restrict__ out_b,
                const unsigned short* __restrict__ gate_t,
                const unsigned short* __restrict__ wp,
                float* __restrict__ out) {
    extern __shared__ char smem[];
    float*          sX   = (float*)smem;                       // [128 c][65 f32] = 33280 B
    unsigned short* sA   = (unsigned short*)(smem + 33280);    // [64 pos][128 c] bf16
    float*          sOut = (float*)smem;                       // reused after LN
    int t = threadIdx.x, wave = t >> 5, lane = t & 31;
    size_t pos0 = (size_t)blockIdx.x * 64;

    if (wave == 0) {
        // data_size=4B (2<<16) | pad_enable (1<<20) | pad_interval=64dw (5<<22)
        tdm_load_2d(0, x_t + pos0, 0x01520000u, NN, CDIM, 64, 128, NN);
        __builtin_amdgcn_s_wait_tensorcnt(0);
    }
    __syncthreads();

    { // LN over 128 channels: 4 lanes per row(pos), 32 channels each
        int row = t >> 2, q = t & 3, ch0 = q * 32;
        float vals[32];
        float s = 0.f, ss = 0.f;
#pragma unroll
        for (int i = 0; i < 32; i++) {
            float v = sX[(ch0 + i) * 65 + row];
            vals[i] = v; s += v; ss += v * v;
        }
        s += __shfl_xor(s, 1, 32); ss += __shfl_xor(ss, 1, 32);
        s += __shfl_xor(s, 2, 32); ss += __shfl_xor(ss, 2, 32);
        float mu   = s * (1.f / CDIM);
        float rstd = rsqrtf(ss * (1.f / CDIM) - mu * mu + 1e-5f);
#pragma unroll
        for (int i = 0; i < 32; i++) {
            sA[row * CDIM + ch0 + i] = f2bf((vals[i] - mu) * rstd * g_c[ch0 + i] + b_c[ch0 + i]);
        }
    }
    __syncthreads();   // sA ready; all sX reads done (sOut may overwrite region)

    int mloc = (wave >> 1) * 16;
    int ntb  = (wave & 1) * 4;
    int lrow = lane & 15, half = lane >> 4;
    for (int nt = ntb; nt < ntb + 4; nt++) {
        v8f ac = {};
#pragma unroll
        for (int kt = 0; kt < 4; kt++) {
            v16bf a = lda(sA, mloc, CDIM, kt * 32, lane);
            ac = wmma_bf16(a, ldw(wp, 5, nt, kt, lane), ac);
        }
        int   n  = nt * 16 + lrow;
        float ob = out_b[n];
        ushort8 g8 = *(const ushort8*)(gate_t + (size_t)n * NN + pos0 + mloc + half * 8);
#pragma unroll
        for (int r = 0; r < 8; r++)
            sOut[(mloc + half * 8 + r) * 129 + n] = (ac[r] + ob) * bf2f(g8[r]);
    }
    __syncthreads();

    for (int it = 0; it < 32; it++) {   // coalesced store
        int idx = it * 256 + t;
        int cc = idx & 127;
        int p  = idx >> 7;
        out[(pos0 + p) * CDIM + cc] = sOut[p * 129 + cc];
    }
}

// ---------- host launcher ----------
extern "C" void kernel_launch(void* const* d_in, const int* in_sizes, int n_in,
                              void* d_out, int out_size, void* d_ws, size_t ws_size,
                              hipStream_t stream) {
    const float* act     = (const float*)d_in[0];
    const float* mask    = (const float*)d_in[1];
    const float* ln_in_g = (const float*)d_in[2];
    const float* ln_in_b = (const float*)d_in[3];
    const float* lp_w    = (const float*)d_in[4];
    const float* lp_b    = (const float*)d_in[5];
    const float* rp_w    = (const float*)d_in[6];
    const float* rp_b    = (const float*)d_in[7];
    const float* lg_w    = (const float*)d_in[8];
    const float* lg_b    = (const float*)d_in[9];
    const float* rg_w    = (const float*)d_in[10];
    const float* rg_b    = (const float*)d_in[11];
    const float* ln_c_g  = (const float*)d_in[12];
    const float* ln_c_b  = (const float*)d_in[13];
    const float* out_w   = (const float*)d_in[14];
    const float* out_b   = (const float*)d_in[15];
    const float* gate_w  = (const float*)d_in[16];
    const float* gate_b  = (const float*)d_in[17];
    float* out = (float*)d_out;

    char* ws = (char*)d_ws;
    unsigned short* wpack   = (unsigned short*)ws;                    // 192KB packed weights
    unsigned short* left_t  = (unsigned short*)(ws + 196608);         // [C][NN] bf16, 64MB
    unsigned short* right_t = left_t  + (size_t)CDIM * NN;            // 64MB
    unsigned short* gate_t  = right_t + (size_t)CDIM * NN;            // 64MB
    float*          x_t     = (float*)(gate_t + (size_t)CDIM * NN);   // [C][NN] f32, 128MB

    pack_w_kernel<<<dim3(8, 4, 6), 32, 0, stream>>>(lp_w, lg_w, rp_w, rg_w, gate_w, out_w, wpack);
    proj_kernel<<<NN / 128, 256, 0, stream>>>(act, mask, ln_in_g, ln_in_b,
                                              lp_b, lg_b, rp_b, rg_b, gate_b,
                                              wpack, left_t, right_t, gate_t);
    tri_kernel<<<dim3(4, 4, CDIM), 256, 32768, stream>>>(left_t, right_t, x_t);
    out_kernel<<<NN / 64, 256, 49664, stream>>>(x_t, ln_c_g, ln_c_b, out_b, gate_t, wpack, out);
}
